// RecurrentGCN_15135464751774
// MI455X (gfx1250) — matrix operand order repaired and checked by reading, not compiled, not run on hardware
//
#include <hip/hip_runtime.h>

// ---------------------------------------------------------------------------
// GConvLSTM (ChebConv K=3, H=64) for MI455X / gfx1250, wave32 + WMMA bf16.
// Roofline: edge gather/scatter dominates (~1.6 GB @ 23.3 TB/s, ~205M f32
// atomics); GEMMs are ~7 GFLOP and run on v_wmma_f32_16x16x32_bf16 (f32 acc).
// Weights are pre-packed into WMMA-fragment-major layout so every B fragment
// is 2x global_load_b128; A fragments are 2x b128 from row-major bf16.
// ---------------------------------------------------------------------------

typedef __attribute__((ext_vector_type(16))) __bf16        v16bf;
typedef __attribute__((ext_vector_type(8)))  float         v8f;
typedef __attribute__((ext_vector_type(4)))  unsigned int  v4u;

#define HDIM 64

// ---------------- small elementwise / graph kernels ----------------

__global__ void k_zero_f32(float* p, long n) {
    long i = (long)blockIdx.x * blockDim.x + threadIdx.x;
    if (i < n) p[i] = 0.0f;
}

__global__ void k_degree(const int* src, const float* ew, float* deg, int E) {
    int e = blockIdx.x * blockDim.x + threadIdx.x;
    if (e < E) atomicAdd(&deg[src[e]], ew[e]);
}

__global__ void k_node_norm(const float* deg, float* dinv, float* diag, int N) {
    int n = blockIdx.x * blockDim.x + threadIdx.x;
    if (n >= N) return;
    float dv = deg[n];
    bool pos = dv > 0.0f;
    dinv[n] = pos ? rsqrtf(dv) : 0.0f;
    diag[n] = pos ? 0.0f : -1.0f;
}

__global__ void k_edge_norm(const int* src, const int* dst, const float* ew,
                            const float* dinv, float* norm, int E) {
    int e = blockIdx.x * blockDim.x + threadIdx.x;
    if (e < E) norm[e] = -dinv[src[e]] * ew[e] * dinv[dst[e]];
}

// scatter:  agg[dst] += norm * z[src]   (thread per edge-feature, coalesced)
__global__ void k_prop_edges(const float* __restrict__ z, const float* __restrict__ norm,
                             const int* __restrict__ src, const int* __restrict__ dst,
                             float* __restrict__ agg, long total) {
    long gid = (long)blockIdx.x * blockDim.x + threadIdx.x;
    if (gid >= total) return;
    int e = (int)(gid >> 6);
    int f = (int)(gid & 63);
    float nv = norm[e];
    int s = src[e], d = dst[e];
    atomicAdd(&agg[(long)d * HDIM + f], nv * z[(long)s * HDIM + f]);
}

// Tx1 = agg + diag * h
__global__ void k_combine1(const float* agg, const float* diag, const float* h,
                           float* tx1, long nh) {
    long i = (long)blockIdx.x * blockDim.x + threadIdx.x;
    if (i >= nh) return;
    tx1[i] = agg[i] + diag[i >> 6] * h[i];
}

// Tx2 = 2*(agg + diag*Tx1) - h   (written in place into agg)
__global__ void k_combine2(float* agg, const float* diag, const float* tx1,
                           const float* h, long nh) {
    long i = (long)blockIdx.x * blockDim.x + threadIdx.x;
    if (i >= nh) return;
    agg[i] = 2.0f * (agg[i] + diag[i >> 6] * tx1[i]) - h[i];
}

__global__ void k_f32_to_bf16(const float* s, __bf16* d, long n) {
    long i = (long)blockIdx.x * blockDim.x + threadIdx.x;
    if (i < n) d[i] = (__bf16)s[i];
}

// Pack the 17 64x64 weight matrices [W_x(4) | theta(12) | lin_w(1)] into
// WMMA-B fragment-major bf16 layout:
//   dst[((m*8 + kb*4 + nb)*32 + lane)*16 + e] = w_m[k*64 + n]
//   k = kb*32 + (lane>>4)*8 + (e&7) + (e>>3)*16 ;  n = nb*16 + (lane&15)
__global__ void k_pack_weights(const float* W_x, const float* theta, const float* lin_w,
                               __bf16* wbp) {
    int i = blockIdx.x * blockDim.x + threadIdx.x;
    if (i >= 17 * 4096) return;
    int m    = i >> 12;          // matrix index
    int r    = i & 4095;
    int frag = r >> 9;           // 0..7  (kb*4+nb)
    int lane = (r >> 4) & 31;
    int e    = r & 15;
    int kb = frag >> 2, nb = frag & 3;
    int k = kb * 32 + (lane >> 4) * 8 + (e & 7) + (e >> 3) * 16;
    int n = nb * 16 + (lane & 15);
    int sidx = k * HDIM + n;
    float v;
    if (m < 4)       v = W_x[m * 4096 + sidx];
    else if (m < 16) v = theta[(m - 4) * 4096 + sidx];
    else             v = lin_w[sidx];
    wbp[i] = (__bf16)v;
}

// ---------------- WMMA fragment helpers (wave32) ----------------

// A fragment: 16x32 bf16 from row-major source. Per lane the 16 elements are
// two contiguous 8-element runs: K = base+{0..7} and base+{16..23}  -> 2x b128.
__device__ inline v16bf load_a_frag(const __bf16* __restrict__ M, int row0, int kb, int maxRow) {
    int lane = threadIdx.x & 31;
    int half = lane >> 4;
    int m = row0 + (lane & 15);
    if (m > maxRow) m = maxRow;
    const __bf16* p = M + (long)m * HDIM + kb * 32 + half * 8;
    union { v16bf v; v4u q[2]; } r;
    r.q[0] = *(const v4u*)(p);
    r.q[1] = *(const v4u*)(p + 16);
    return r.v;
}

// B fragment from pre-packed weights: two 8-element b128 runs per lane, at p and p+16 (next lane's slot).
__device__ inline v16bf load_b_frag(const __bf16* __restrict__ Wp, int kb, int nb) {
    int lane = threadIdx.x & 31;
    const __bf16* p = Wp + (((kb * 4 + nb) * 32 + lane) << 4);
    union { v16bf v; v4u q[2]; } r;
    r.q[0] = *(const v4u*)(p);
    r.q[1] = *(const v4u*)(p + 16);
    return r.v;
}

// D store: v8f -> rows row0 + half*8 + r, col = nb*16 + lane&15
__device__ inline void store_d(float* __restrict__ C, int row0, int nb, v8f acc, int N) {
    int lane = threadIdx.x & 31;
    int half = lane >> 4;
    int col = nb * 16 + (lane & 15);
#pragma unroll
    for (int r = 0; r < 8; ++r) {
        int row = row0 + half * 8 + r;
        if (row < N) C[(long)row * HDIM + col] = acc[r];
    }
}

__device__ inline v8f wmma_bf16(v16bf a, v16bf b, v8f c) {
    return __builtin_amdgcn_wmma_f32_16x16x32_bf16(
        /*neg_a=*/false, a, /*neg_b=*/false, b,
        /*c_mod=*/(short)0, c, /*reuse_a=*/false, /*reuse_b=*/false);
}

// ---------------- fused 4-gate GEMM ----------------
// gates[g] = x@W_x[g] + h@th[g,0] + Tx1@th[g,1] + Tx2@th[g,2] + conv_b[g] + b[g]
__global__ void __launch_bounds__(32)
k_gates_gemm(const __bf16* __restrict__ xb, const __bf16* __restrict__ hb,
             const __bf16* __restrict__ t1b, const __bf16* __restrict__ t2b,
             const __bf16* __restrict__ wbp,
             const float* __restrict__ conv_b, const float* __restrict__ bias,
             float* __restrict__ gates, int N) {
    int row0 = blockIdx.x * 16;
    if (row0 >= N) return;
    int lane = threadIdx.x & 31;
    int maxRow = N - 1;

    v16bf ax0 = load_a_frag(xb,  row0, 0, maxRow), ax1 = load_a_frag(xb,  row0, 1, maxRow);
    v16bf ah0 = load_a_frag(hb,  row0, 0, maxRow), ah1 = load_a_frag(hb,  row0, 1, maxRow);
    v16bf a10 = load_a_frag(t1b, row0, 0, maxRow), a11 = load_a_frag(t1b, row0, 1, maxRow);
    v16bf a20 = load_a_frag(t2b, row0, 0, maxRow), a21 = load_a_frag(t2b, row0, 1, maxRow);

    long planeStride = (long)N * HDIM;
    for (int g = 0; g < 4; ++g) {
        const __bf16* wx  = wbp + (long)g * 4096;
        const __bf16* th0 = wbp + (long)(4 + g * 3 + 0) * 4096;
        const __bf16* th1 = wbp + (long)(4 + g * 3 + 1) * 4096;
        const __bf16* th2 = wbp + (long)(4 + g * 3 + 2) * 4096;
        float* out = gates + g * planeStride;
#pragma unroll
        for (int nb = 0; nb < 4; ++nb) {
            int col = nb * 16 + (lane & 15);
            float bv = conv_b[g * HDIM + col] + bias[g * HDIM + col];
            v8f acc;
#pragma unroll
            for (int r = 0; r < 8; ++r) acc[r] = bv;
            acc = wmma_bf16(ax0, load_b_frag(wx,  0, nb), acc);
            acc = wmma_bf16(ax1, load_b_frag(wx,  1, nb), acc);
            acc = wmma_bf16(ah0, load_b_frag(th0, 0, nb), acc);
            acc = wmma_bf16(ah1, load_b_frag(th0, 1, nb), acc);
            acc = wmma_bf16(a10, load_b_frag(th1, 0, nb), acc);
            acc = wmma_bf16(a11, load_b_frag(th1, 1, nb), acc);
            acc = wmma_bf16(a20, load_b_frag(th2, 0, nb), acc);
            acc = wmma_bf16(a21, load_b_frag(th2, 1, nb), acc);
            store_d(out, row0, nb, acc, N);
        }
    }
}

// ---------------- LSTM gate elementwise ----------------
__device__ inline float sigmf(float x) { return 1.0f / (1.0f + __expf(-x)); }

__global__ void k_gate_eltwise(const float* __restrict__ gates, const float* __restrict__ c,
                               const float* __restrict__ w_c,
                               float* __restrict__ outHn, float* __restrict__ outCn,
                               __bf16* __restrict__ relub, long nh) {
    long i = (long)blockIdx.x * blockDim.x + threadIdx.x;
    if (i >= nh) return;
    int f = (int)(i & 63);
    float A0 = gates[i], A1 = gates[nh + i], A2 = gates[2 * nh + i], A3 = gates[3 * nh + i];
    float cv = c[i];
    float I  = sigmf(A0 + w_c[f] * cv);
    float Fg = sigmf(A1 + w_c[64 + f] * cv);
    float T  = tanhf(A2);
    float Cn = Fg * cv + I * T;
    float O  = sigmf(A3 + w_c[128 + f] * Cn);
    float Hn = O * tanhf(Cn);
    outHn[i] = Hn;
    outCn[i] = Cn;
    relub[i] = (__bf16)(Hn > 0.0f ? Hn : 0.0f);
}

// ---------------- final linear: h_out = relu(Hn) @ lin_w + lin_b ----------------
__global__ void __launch_bounds__(32)
k_lin_gemm(const __bf16* __restrict__ relub, const __bf16* __restrict__ lwp,
           const float* __restrict__ lin_b, float* __restrict__ out, int N) {
    int row0 = blockIdx.x * 16;
    if (row0 >= N) return;
    int lane = threadIdx.x & 31;
    int maxRow = N - 1;
    v16bf a0 = load_a_frag(relub, row0, 0, maxRow);
    v16bf a1 = load_a_frag(relub, row0, 1, maxRow);
#pragma unroll
    for (int nb = 0; nb < 4; ++nb) {
        int col = nb * 16 + (lane & 15);
        float bv = lin_b[col];
        v8f acc;
#pragma unroll
        for (int r = 0; r < 8; ++r) acc[r] = bv;
        acc = wmma_bf16(a0, load_b_frag(lwp, 0, nb), acc);
        acc = wmma_bf16(a1, load_b_frag(lwp, 1, nb), acc);
        store_d(out, row0, nb, acc, N);
    }
}

// ---------------- host orchestration ----------------

extern "C" void kernel_launch(void* const* d_in, const int* in_sizes, int n_in,
                              void* d_out, int out_size, void* d_ws, size_t ws_size,
                              hipStream_t stream) {
    const float* x      = (const float*)d_in[0];
    const int*   eidx   = (const int*)  d_in[1];
    const float* ew     = (const float*)d_in[2];
    const float* h      = (const float*)d_in[3];
    const float* c      = (const float*)d_in[4];
    const float* W_x    = (const float*)d_in[5];
    const float* w_c    = (const float*)d_in[6];
    const float* b      = (const float*)d_in[7];
    const float* theta  = (const float*)d_in[8];
    const float* conv_b = (const float*)d_in[9];
    const float* lin_w  = (const float*)d_in[10];
    const float* lin_b  = (const float*)d_in[11];

    const int N = in_sizes[0] / HDIM;
    const int E = in_sizes[2];
    const long NH = (long)N * HDIM;
    const int* src = eidx;
    const int* dst = eidx + E;

    // workspace carve-out
    char* wsb = (char*)d_ws;
    size_t o = 0;
    auto alloc = [&](size_t bytes) -> void* {
        o = (o + 255) & ~(size_t)255;
        void* p = wsb + o;
        o += bytes;
        return p;
    };
    float* deg   = (float*)alloc((size_t)N * 4);
    float* dinv  = (float*)alloc((size_t)N * 4);
    float* diag  = (float*)alloc((size_t)N * 4);
    float* norm  = (float*)alloc((size_t)E * 4);
    float* tx1   = (float*)alloc((size_t)NH * 4);
    float* agg   = (float*)alloc((size_t)NH * 4);      // also holds Tx2 after combine2
    float* gates = (float*)alloc((size_t)4 * NH * 4);
    __bf16* xb    = (__bf16*)alloc((size_t)NH * 2);
    __bf16* hb    = (__bf16*)alloc((size_t)NH * 2);
    __bf16* t1b   = (__bf16*)alloc((size_t)NH * 2);
    __bf16* t2b   = (__bf16*)alloc((size_t)NH * 2);
    __bf16* relub = (__bf16*)alloc((size_t)NH * 2);
    __bf16* wbp   = (__bf16*)alloc((size_t)17 * 4096 * 2);
    (void)ws_size; (void)n_in;

    float* outH  = (float*)d_out;                  // h_out
    float* outHn = (float*)d_out + NH;             // Hn
    float* outCn = (float*)d_out + 2 * NH;         // Cn
    (void)out_size;

    const int B = 256;
    long totalEF = (long)E * HDIM;
    int gN   = (int)((N + B - 1) / B);
    int gE   = (int)((E + B - 1) / B);
    int gNH  = (int)((NH + B - 1) / B);
    int gEF  = (int)((totalEF + B - 1) / B);
    int gTile = (N + 15) / 16;

    // 1. graph normalization
    k_zero_f32<<<gN, B, 0, stream>>>(deg, N);
    k_degree<<<gE, B, 0, stream>>>(src, ew, deg, E);
    k_node_norm<<<gN, B, 0, stream>>>(deg, dinv, diag, N);
    k_edge_norm<<<gE, B, 0, stream>>>(src, dst, ew, dinv, norm, E);

    // 2. weight / activation packing to bf16 (weights -> fragment-major)
    k_pack_weights<<<(17 * 4096 + B - 1) / B, B, 0, stream>>>(W_x, theta, lin_w, wbp);
    k_f32_to_bf16<<<gNH, B, 0, stream>>>(x, xb, NH);
    k_f32_to_bf16<<<gNH, B, 0, stream>>>(h, hb, NH);

    // 3. Chebyshev basis: Tx1 = prop(h)
    k_zero_f32<<<gNH, B, 0, stream>>>(agg, NH);
    k_prop_edges<<<gEF, B, 0, stream>>>(h, norm, src, dst, agg, totalEF);
    k_combine1<<<gNH, B, 0, stream>>>(agg, diag, h, tx1, NH);
    k_f32_to_bf16<<<gNH, B, 0, stream>>>(tx1, t1b, NH);

    //    Tx2 = 2*prop(Tx1) - h   (in place in agg)
    k_zero_f32<<<gNH, B, 0, stream>>>(agg, NH);
    k_prop_edges<<<gEF, B, 0, stream>>>(tx1, norm, src, dst, agg, totalEF);
    k_combine2<<<gNH, B, 0, stream>>>(agg, diag, tx1, h, NH);
    k_f32_to_bf16<<<gNH, B, 0, stream>>>(agg, t2b, NH);

    // 4. fused 4-gate GEMM (WMMA bf16 -> f32)
    k_gates_gemm<<<gTile, 32, 0, stream>>>(xb, hb, t1b, t2b, wbp, conv_b, b, gates, N);

    // 5. LSTM elementwise
    k_gate_eltwise<<<gNH, B, 0, stream>>>(gates, c, w_c, outHn, outCn, relub, NH);

    // 6. output projection
    k_lin_gemm<<<gTile, 32, 0, stream>>>(relub, wbp + (long)16 * 4096, lin_b, outH, N);
}
